// ViMHA_5978594476209
// MI455X (gfx1250) — matrix operand order repaired and checked by reading, not compiled, not run on hardware
//
#include <hip/hip_runtime.h>
#include <hip/hip_bf16.h>

typedef __attribute__((ext_vector_type(16))) _Float16 v16h;
typedef __attribute__((ext_vector_type(8)))  _Float16 v8h;
typedef __attribute__((ext_vector_type(8)))  float    v8f;

#define WMMA16(a, b, c) __builtin_amdgcn_wmma_f32_16x16x32_f16( \
    false, (a), false, (b), (short)0, (c), false, false)

// Problem constants: B=16, C=3, H=W=512, P=16 -> nP=1024, D=768, nh=12, dh=64
#define NB   16
#define NP   1024
#define DIM  768
#define NH   12
#define DH   64

__device__ __forceinline__ v16h make_v16(v8h lo, v8h hi) {
  v16h r;
#pragma unroll
  for (int i = 0; i < 8; ++i) { r[i] = lo[i]; r[i + 8] = hi[i]; }
  return r;
}

// ---------------------------------------------------------------------------
// prep_w: W (nh,768,64) f32 -> Wt (nh,64,768) f16  (transposed for B-fragments)
// ---------------------------------------------------------------------------
__global__ void prep_w(const float* __restrict__ Wq, const float* __restrict__ Wk,
                       const float* __restrict__ Wv,
                       _Float16* __restrict__ wqt, _Float16* __restrict__ wkt,
                       _Float16* __restrict__ wvt) {
  int idx = blockIdx.x * blockDim.x + threadIdx.x;
  const int per = NH * DIM * DH;  // 589824
  if (idx >= 3 * per) return;
  int mat = idx / per;
  int r = idx % per;
  int h = r / (DIM * DH);
  int r2 = r % (DIM * DH);
  int k = r2 / DH;
  int n = r2 % DH;
  const float* src = (mat == 0) ? Wq : (mat == 1) ? Wk : Wv;
  _Float16* dst = (mat == 0) ? wqt : (mat == 1) ? wkt : wvt;
  dst[((size_t)h * DH + n) * DIM + k] = (_Float16)src[((size_t)h * DIM + k) * DH + n];
}

// ---------------------------------------------------------------------------
// prep_t: tokenize (torch .view interleave) + sinusoidal PE, T = 2x + pe (f16)
// ---------------------------------------------------------------------------
__global__ void prep_t(const float* __restrict__ x, _Float16* __restrict__ t16) {
  int idx = blockIdx.x * blockDim.x + threadIdx.x;
  const int perB = NP * DIM;  // 786432 = 2^18 * 3
  if (idx >= NB * perB) return;
  int b = idx / perB;
  int f = idx % perB;
  int n = f / DIM;
  int d = f % DIM;
  // .view interleave: f = c*2^18 + patch*256 + py*16 + px (nH=nW=32)
  int c = f >> 18;
  int rem = f & 262143;
  int patch = rem >> 8;
  int pix = rem & 255;
  int row = ((patch >> 5) << 4) + (pix >> 4);
  int col = ((patch & 31) << 4) + (pix & 15);
  float xv = x[(((size_t)(b * 3 + c) * 512) + row) * 512 + col];
  int i2 = (d >> 1) << 1;  // 2*(d//2)
  float ang = (float)n * __expf(-(float)i2 * (9.210340371976184f / (float)DIM));
  float pe = (d & 1) ? __cosf(ang) : __sinf(ang);
  t16[idx] = (_Float16)(2.0f * xv + pe);
}

// ---------------------------------------------------------------------------
// qkv_kernel: one wave -> 16 tokens x 64 cols of one head of Q/K/V via WMMA.
//   Q16,K16: (B,nh,1024,64) f16 (Q pre-scaled by 1/8)
//   V16t:    (B,nh,64,1024) f16 (transposed for PV B-fragments)
// ---------------------------------------------------------------------------
__global__ void qkv_kernel(const _Float16* __restrict__ t16,
                           const _Float16* __restrict__ wqt,
                           const _Float16* __restrict__ wkt,
                           const _Float16* __restrict__ wvt,
                           const float* __restrict__ bq,
                           const float* __restrict__ bk,
                           const float* __restrict__ bv,
                           _Float16* __restrict__ q16,
                           _Float16* __restrict__ k16,
                           _Float16* __restrict__ v16t) {
  const int lane = threadIdx.x & 31;
  const int wave = threadIdx.x >> 5;
  const int waveId = blockIdx.x * 8 + wave;
  const int mtile = waveId & 63;          // 64 tiles of 16 tokens
  int tmp = waveId >> 6;
  const int h = tmp % NH; tmp /= NH;
  const int b = tmp & 15;
  const int mat = tmp >> 4;               // 0=Q 1=K 2=V

  const int ln = lane & 15;
  const int hi = lane >> 4;

  const _Float16* W = (mat == 0) ? wqt : (mat == 1) ? wkt : wvt;
  const float* bias = (mat == 0) ? bq : (mat == 1) ? bk : bv;

  const _Float16* arow = t16 + ((size_t)(b * NP + mtile * 16 + ln)) * DIM;
  const _Float16* wbase = W + (size_t)h * DH * DIM;

  v8f acc[4] = {};
#pragma unroll 4
  for (int k0 = 0; k0 < DIM; k0 += 32) {
    v16h a = make_v16(*(const v8h*)(arow + k0 + hi * 8),
                      *(const v8h*)(arow + k0 + 16 + hi * 8));
#pragma unroll
    for (int t = 0; t < 4; ++t) {
      const _Float16* wr = wbase + (size_t)(t * 16 + ln) * DIM + k0 + hi * 8;
      v16h bf = make_v16(*(const v8h*)wr, *(const v8h*)(wr + 16));
      acc[t] = WMMA16(a, bf, acc[t]);
    }
  }
  // bias (per output column)
#pragma unroll
  for (int t = 0; t < 4; ++t) {
    float bb = bias[h * DH + t * 16 + ln];
#pragma unroll
    for (int r = 0; r < 8; ++r) acc[t][r] += bb;
  }

  if (mat == 2) {
    // transposed store: rows m = r + 8*hi are consecutive kv tokens -> b128
#pragma unroll
    for (int t = 0; t < 4; ++t) {
      v8h pk;
#pragma unroll
      for (int r = 0; r < 8; ++r) pk[r] = (_Float16)acc[t][r];
      _Float16* dst = v16t +
          ((size_t)((b * NH + h) * DH + t * 16 + ln)) * NP + mtile * 16 + hi * 8;
      *(v8h*)dst = pk;
    }
  } else {
    _Float16* Out = (mat == 0) ? q16 : k16;
    const float scl = (mat == 0) ? 0.125f : 1.0f;  // fold 1/sqrt(dh) into Q
#pragma unroll
    for (int t = 0; t < 4; ++t) {
#pragma unroll
      for (int r = 0; r < 8; ++r) {
        int token = mtile * 16 + r + 8 * hi;
        Out[((size_t)((b * NH + h) * NP + token)) * DH + t * 16 + ln] =
            (_Float16)(acc[t][r] * scl);
      }
    }
  }
}

// ---------------------------------------------------------------------------
// attn_kernel: flash attention. wave = 16 query rows x full dh=64.
// block = 8 waves = 128 query rows; grid = B*nh*8.
// ---------------------------------------------------------------------------
__global__ void attn_kernel(const _Float16* __restrict__ q16,
                            const _Float16* __restrict__ k16,
                            const _Float16* __restrict__ v16t,
                            float* __restrict__ out) {
  __shared__ __align__(16) _Float16 pbuf[8][16 * 32];
  const int lane = threadIdx.x & 31;
  const int wave = threadIdx.x >> 5;
  const int ln = lane & 15;
  const int hi = lane >> 4;
  const int qt = blockIdx.x & 7;
  int bh = blockIdx.x >> 3;
  const int h = bh % NH;
  const int b = bh / NH;
  const int q0 = qt * 128 + wave * 16;

  // Q A-fragments for dh chunks 0..31, 32..63 (held in registers all loop)
  const _Float16* qrow = q16 + ((size_t)((b * NH + h) * NP + q0 + ln)) * DH;
  v16h qa0 = make_v16(*(const v8h*)(qrow + hi * 8),
                      *(const v8h*)(qrow + 16 + hi * 8));
  v16h qa1 = make_v16(*(const v8h*)(qrow + 32 + hi * 8),
                      *(const v8h*)(qrow + 48 + hi * 8));

  const _Float16* kbase = k16 + (size_t)(b * NH + h) * NP * DH;
  const _Float16* vbase = v16t + (size_t)(b * NH + h) * DH * NP;

  v8f o[4] = {};
  float mrow[8], lrow[8];
#pragma unroll
  for (int r = 0; r < 8; ++r) { mrow[r] = -3.0e38f; lrow[r] = 0.0f; }

  _Float16* pl = &pbuf[wave][0];

  for (int kv0 = 0; kv0 < NP; kv0 += 32) {
    // S = Q * K^T (scale folded into Q). B-frag: lane holds kv row -> contiguous
    const _Float16* kr0 = kbase + (size_t)(kv0 + ln) * DH;
    const _Float16* kr1 = kbase + (size_t)(kv0 + 16 + ln) * DH;
    v16h bk00 = make_v16(*(const v8h*)(kr0 + hi * 8), *(const v8h*)(kr0 + 16 + hi * 8));
    v16h bk01 = make_v16(*(const v8h*)(kr0 + 32 + hi * 8), *(const v8h*)(kr0 + 48 + hi * 8));
    v16h bk10 = make_v16(*(const v8h*)(kr1 + hi * 8), *(const v8h*)(kr1 + 16 + hi * 8));
    v16h bk11 = make_v16(*(const v8h*)(kr1 + 32 + hi * 8), *(const v8h*)(kr1 + 48 + hi * 8));
    v8f s0 = {};
    s0 = WMMA16(qa0, bk00, s0);
    s0 = WMMA16(qa1, bk01, s0);
    v8f s1 = {};
    s1 = WMMA16(qa0, bk10, s1);
    s1 = WMMA16(qa1, bk11, s1);

    // online softmax; C-fragment row m = r + 8*hi lives in one 16-lane half
#pragma unroll
    for (int r = 0; r < 8; ++r) {
      float vmax = fmaxf(s0[r], s1[r]);
      vmax = fmaxf(vmax, __shfl_xor(vmax, 1));
      vmax = fmaxf(vmax, __shfl_xor(vmax, 2));
      vmax = fmaxf(vmax, __shfl_xor(vmax, 4));
      vmax = fmaxf(vmax, __shfl_xor(vmax, 8));
      float mnew = fmaxf(mrow[r], vmax);
      float alpha = __expf(mrow[r] - mnew);
      mrow[r] = mnew;
      float p0 = __expf(s0[r] - mnew);
      float p1 = __expf(s1[r] - mnew);
      float rs = p0 + p1;
      rs += __shfl_xor(rs, 1);
      rs += __shfl_xor(rs, 2);
      rs += __shfl_xor(rs, 4);
      rs += __shfl_xor(rs, 8);
      lrow[r] = lrow[r] * alpha + rs;
#pragma unroll
      for (int t = 0; t < 4; ++t) o[t][r] *= alpha;
      const int m = r + 8 * hi;
      pl[m * 32 + ln] = (_Float16)p0;
      pl[m * 32 + 16 + ln] = (_Float16)p1;
    }
    // within-wave LDS store->load ordering (CDNA5 split DS counter)
    asm volatile("s_wait_dscnt 0" ::: "memory");

    // P as A-fragment (C-layout -> A-layout transpose through LDS)
    const _Float16* prow = pl + ln * 32;
    v16h pa = make_v16(*(const v8h*)(prow + hi * 8),
                       *(const v8h*)(prow + 16 + hi * 8));

    // O += P * V (V stored transposed: lane holds dh column -> contiguous kv)
#pragma unroll
    for (int t = 0; t < 4; ++t) {
      const _Float16* vr = vbase + (size_t)(t * 16 + ln) * NP + kv0 + hi * 8;
      v16h bvf = make_v16(*(const v8h*)vr, *(const v8h*)(vr + 16));
      o[t] = WMMA16(pa, bvf, o[t]);
    }
  }

  // normalize + scatter through the inverse .view/untokenize mapping
#pragma unroll
  for (int t = 0; t < 4; ++t) {
#pragma unroll
    for (int r = 0; r < 8; ++r) {
      float val = o[t][r] / lrow[r];
      int token = q0 + r + 8 * hi;
      int j = h * DH + t * 16 + ln;
      int fo = token * DIM + j;
      int c = fo >> 18;
      int rem = fo & 262143;
      int patch = rem >> 8;
      int pix = rem & 255;
      int row = ((patch >> 5) << 4) + (pix >> 4);
      int col = ((patch & 31) << 4) + (pix & 15);
      out[(((size_t)(b * 3 + c) * 512) + row) * 512 + col] = val;
    }
  }
}

// ---------------------------------------------------------------------------
extern "C" void kernel_launch(void* const* d_in, const int* in_sizes, int n_in,
                              void* d_out, int out_size, void* d_ws, size_t ws_size,
                              hipStream_t stream) {
  (void)in_sizes; (void)n_in; (void)out_size; (void)ws_size;
  const float* x  = (const float*)d_in[0];
  const float* Wq = (const float*)d_in[1];
  const float* bq = (const float*)d_in[2];
  const float* Wk = (const float*)d_in[3];
  const float* bk = (const float*)d_in[4];
  const float* Wv = (const float*)d_in[5];
  const float* bv = (const float*)d_in[6];
  float* out = (float*)d_out;

  _Float16* ws = (_Float16*)d_ws;
  const size_t WT = (size_t)NH * DH * DIM;     // 589,824 halves each
  const size_t TQ = (size_t)NB * NP * DIM;     // 12,582,912 halves (= B*nh*nP*dh)
  _Float16* wqt = ws;
  _Float16* wkt = wqt + WT;
  _Float16* wvt = wkt + WT;
  _Float16* t16 = wvt + WT;
  _Float16* q16 = t16 + TQ;
  _Float16* k16 = q16 + TQ;
  _Float16* v16t = k16 + TQ;
  // total ws use: 3*WT + 4*TQ halves  ~= 104 MB

  prep_w<<<(3 * NH * DIM * DH + 255) / 256, 256, 0, stream>>>(Wq, Wk, Wv, wqt, wkt, wvt);
  prep_t<<<((int)TQ + 255) / 256, 256, 0, stream>>>(x, t16);
  // waves: 3 mats * 16 B * 12 h * 64 mtiles = 36864 -> 4608 blocks of 8 waves
  qkv_kernel<<<4608, 256, 0, stream>>>(t16, wqt, wkt, wvt, bq, bk, bv, q16, k16, v16t);
  // blocks: B*nh*8 qtiles = 1536
  attn_kernel<<<1536, 256, 0, stream>>>(q16, k16, v16t, out);
}
